// DynamicScatterVFE_31550829757176
// MI455X (gfx1250) — matrix operand order repaired and checked by reading, not compiled, not run on hardware
//
#include <hip/hip_runtime.h>
#include <stdint.h>

typedef _Float16 h16_t;
typedef __attribute__((ext_vector_type(16))) _Float16 v16h;
typedef __attribute__((ext_vector_type(8)))  _Float16 v8h;
typedef __attribute__((ext_vector_type(8)))  float    v8f;

#define BN_EPS 1e-3f
// voxel geometry constants from the reference
#define C_VX 0.2f
#define C_VY 0.2f
#define C_VZ 4.0f
#define C_XOFF 0.1f      /* VX/2 + 0.0  */
#define C_YOFF (-39.9f)  /* VY/2 - 40.0 */
#define C_ZOFF (-1.0f)   /* VZ/2 - 3.0  */

// ---------------------------------------------------------------------------
// Weight pre-shuffle: fragment-major f16 layout so each lane's WMMA B operand
// is one contiguous 32-byte load.
// B layout (16x16x32 f16): lane n<16 -> col N=n, K=kbase+h (h<8), kbase+16+(h-8) (h>=8);
//                          lane>=16 -> col N=n-16, K=kbase+8+h (h<8), kbase+24+(h-8) (h>=8).
// ---------------------------------------------------------------------------
__global__ void k_prep_weights(const float* __restrict__ w0,
                               const float* __restrict__ w1,
                               h16_t* __restrict__ w0frag,
                               h16_t* __restrict__ w1frag) {
  // w0: [10,64] -> 4 fragments (nt=0..3), K padded 10->32 with zeros
  for (int idx = threadIdx.x; idx < 4 * 512; idx += blockDim.x) {
    int frag = idx >> 9, rem = idx & 511, lane = rem >> 4, h = rem & 15;
    int K = ((lane & 16) ? 8 : 0) + (h & 7) + ((h & 8) ? 16 : 0);
    int Nc = frag * 16 + (lane & 15);
    w0frag[idx] = (K < 10) ? (h16_t)w0[K * 64 + Nc] : (h16_t)0.0f;
  }
  // w1: [128,128] -> 32 fragments (nt=0..7, kt=0..3)
  for (int idx = threadIdx.x; idx < 32 * 512; idx += blockDim.x) {
    int frag = idx >> 9, rem = idx & 511, lane = rem >> 4, h = rem & 15;
    int nt = frag >> 2, kt = frag & 3;
    int K = kt * 32 + ((lane & 16) ? 8 : 0) + (h & 7) + ((h & 8) ? 16 : 0);
    int Nc = nt * 16 + (lane & 15);
    w1frag[idx] = (h16_t)w1[K * 128 + Nc];
  }
}

// ---------------------------------------------------------------------------
// Scatter-sum of point features per voxel (for the cluster mean).
// ---------------------------------------------------------------------------
__global__ void k_scatter_sum(const float* __restrict__ feat,
                              const int* __restrict__ inv,
                              float* __restrict__ vsum,
                              float* __restrict__ cnt, int n) {
  int i = blockIdx.x * blockDim.x + threadIdx.x;
  if (i >= n) return;
  int v = inv[i];
#pragma unroll
  for (int c = 0; c < 4; ++c) atomicAdd(&vsum[v * 4 + c], feat[i * 4 + c]);
  atomicAdd(&cnt[v], 1.0f);
}

// ---------------------------------------------------------------------------
// Build 10-feature augmented points, store as f16 [N,16] (K-padded).
// ---------------------------------------------------------------------------
__global__ void k_build_x10(const float* __restrict__ feat,
                            const int* __restrict__ coors,
                            const int* __restrict__ inv,
                            const float* __restrict__ vsum,
                            const float* __restrict__ cnt,
                            h16_t* __restrict__ x10, int n) {
  int i = blockIdx.x * blockDim.x + threadIdx.x;
  if (i >= n) return;
  int v = inv[i];
  float c = fmaxf(cnt[v], 1.0f);
  float f0 = feat[i * 4 + 0], f1 = feat[i * 4 + 1];
  float f2 = feat[i * 4 + 2], f3 = feat[i * 4 + 3];
  float m0 = vsum[v * 4 + 0] / c, m1 = vsum[v * 4 + 1] / c, m2 = vsum[v * 4 + 2] / c;
  float cz = (float)coors[i * 4 + 1];
  float cy = (float)coors[i * 4 + 2];
  float cx = (float)coors[i * 4 + 3];
  union { v8h r[2]; h16_t h[16]; } u;
  u.h[0] = (h16_t)f0; u.h[1] = (h16_t)f1; u.h[2] = (h16_t)f2; u.h[3] = (h16_t)f3;
  u.h[4] = (h16_t)(f0 - m0); u.h[5] = (h16_t)(f1 - m1); u.h[6] = (h16_t)(f2 - m2);
  u.h[7] = (h16_t)(f0 - (cx * C_VX + C_XOFF));
  u.h[8] = (h16_t)(f1 - (cy * C_VY + C_YOFF));
  u.h[9] = (h16_t)(f2 - (cz * C_VZ + C_ZOFF));
#pragma unroll
  for (int k = 10; k < 16; ++k) u.h[k] = (h16_t)0.0f;
  *(v8h*)(x10 + (size_t)i * 16)     = u.r[0];
  *(v8h*)(x10 + (size_t)i * 16 + 8) = u.r[1];
}

// ---------------------------------------------------------------------------
// A-fragment helpers (ISA 16-bit A 16x32 layout):
//   lane<16 : halves 0..7 -> K=kbase..kbase+7 ; halves 8..15 -> K=kbase+16..+23
//   lane>=16: halves 0..7 -> K=kbase+8..+15  ; halves 8..15 -> K=kbase+24..+31
// i.e. two contiguous 8-half runs starting at kbase+(hi?8:0) and +16.
// ---------------------------------------------------------------------------
__device__ inline v16h load_a_x10(const h16_t* __restrict__ x10, int row, int lane) {
  union { v16h v; v8h r[2]; } a;
  a.r[0] = *(const v8h*)(x10 + (size_t)row * 16 + ((lane & 16) ? 8 : 0));
  v8h z = {};
  a.r[1] = z;  // K=16..31 are zero padding
  return a.v;
}

__device__ inline v8h load8_x1(const h16_t* __restrict__ p0row,
                               const float* __restrict__ v0row, int s) {
  if (s < 64) return *(const v8h*)(p0row + s);
  v8f f = *(const v8f*)(v0row + (s - 64));
  v8h r;
#pragma unroll
  for (int i = 0; i < 8; ++i) r[i] = (h16_t)f[i];
  return r;
}

// ---------------------------------------------------------------------------
// GEMM0 pass 1: y0 = x10 @ w0, accumulate per-column sum / sumsq (BN stats).
// One wave per 16-row block; 4 column tiles of 16.
// ---------------------------------------------------------------------------
__global__ __launch_bounds__(256) void k_gemm0_stats(
    const h16_t* __restrict__ x10, const h16_t* __restrict__ w0frag,
    float* __restrict__ sums, int n) {
  __shared__ float s[128];
  int t = threadIdx.x;
  if (t < 128) s[t] = 0.0f;
  __syncthreads();
  int lane = t & 31;
  int rb = blockIdx.x * 8 + (t >> 5);
  if (rb * 16 < n) {
    int row = rb * 16 + (lane & 15);
    v16h a = load_a_x10(x10, row, lane);
    int col0 = lane & 15;
#pragma unroll
    for (int nt = 0; nt < 4; ++nt) {
      v16h b = *(const v16h*)(w0frag + ((size_t)nt * 32 + lane) * 16);
      v8f c = {};
      c = __builtin_amdgcn_wmma_f32_16x16x32_f16(false, a, false, b,
                                                 (short)0, c, false, false);
      float ps = 0.0f, pq = 0.0f;
#pragma unroll
      for (int j = 0; j < 8; ++j) { ps += c[j]; pq += c[j] * c[j]; }
      int col = nt * 16 + col0;
      atomicAdd(&s[col], ps);
      atomicAdd(&s[64 + col], pq);
    }
  }
  __syncthreads();
  if (t < 128) atomicAdd(&sums[t], s[t]);
}

// ---------------------------------------------------------------------------
// Fold BN stats + gamma/beta into per-column scale/shift.
// ---------------------------------------------------------------------------
__global__ void k_finalize_bn(const float* __restrict__ sums,
                              const float* __restrict__ g,
                              const float* __restrict__ b,
                              float* __restrict__ scale,
                              float* __restrict__ shift, int C, float invN) {
  int c = threadIdx.x;
  if (c >= C) return;
  float mean = sums[c] * invN;
  float var = sums[C + c] * invN - mean * mean;
  float is = rsqrtf(var + BN_EPS);
  float sc = g[c] * is;
  scale[c] = sc;
  shift[c] = b[c] - mean * sc;
}

// ---------------------------------------------------------------------------
// GEMM0 pass 2: recompute y0, apply BN+ReLU -> p0 (f16), atomic segment-max
// into v0 (f32, values >= 0 so int-bitpattern max == float max).
// C/D layout: lane holds col N=lane&15, rows M=j+((lane&16)?8:0).
// ---------------------------------------------------------------------------
__global__ __launch_bounds__(256) void k_gemm0_apply(
    const h16_t* __restrict__ x10, const h16_t* __restrict__ w0frag,
    const float* __restrict__ scale, const float* __restrict__ shift,
    const int* __restrict__ inv, h16_t* __restrict__ p0,
    float* __restrict__ v0, int n) {
  int t = threadIdx.x;
  int lane = t & 31;
  int rb = blockIdx.x * 8 + (t >> 5);
  if (rb * 16 >= n) return;
  int row = rb * 16 + (lane & 15);
  v16h a = load_a_x10(x10, row, lane);
  int hi = (lane & 16) ? 8 : 0;
  int vox[8];
#pragma unroll
  for (int j = 0; j < 8; ++j) vox[j] = inv[rb * 16 + hi + j];
  int col0 = lane & 15;
#pragma unroll
  for (int nt = 0; nt < 4; ++nt) {
    v16h b = *(const v16h*)(w0frag + ((size_t)nt * 32 + lane) * 16);
    v8f c = {};
    c = __builtin_amdgcn_wmma_f32_16x16x32_f16(false, a, false, b,
                                               (short)0, c, false, false);
    int col = nt * 16 + col0;
    float sc = scale[col], sh = shift[col];
#pragma unroll
    for (int j = 0; j < 8; ++j) {
      float val = fmaxf(c[j] * sc + sh, 0.0f);
      int M = rb * 16 + hi + j;
      p0[(size_t)M * 64 + col] = (h16_t)val;
      atomicMax((int*)&v0[(size_t)vox[j] * 64 + col], __float_as_int(val));
    }
  }
}

// ---------------------------------------------------------------------------
// GEMM1 pass 1: y1 = [p0 | v0[inv]] @ w1, accumulate BN stats (128 cols).
// ---------------------------------------------------------------------------
__global__ __launch_bounds__(256) void k_gemm1_stats(
    const h16_t* __restrict__ p0, const float* __restrict__ v0,
    const int* __restrict__ inv, const h16_t* __restrict__ w1frag,
    float* __restrict__ sums, int n) {
  __shared__ float s[256];
  int t = threadIdx.x;
  s[t] = 0.0f;
  __syncthreads();
  int lane = t & 31;
  int rb = blockIdx.x * 8 + (t >> 5);
  if (rb * 16 < n) {
    int row = rb * 16 + (lane & 15);
    int vx = inv[row];
    const h16_t* p0row = p0 + (size_t)row * 64;
    const float* v0row = v0 + (size_t)vx * 64;
    v16h a[4];
#pragma unroll
    for (int kt = 0; kt < 4; ++kt) {
      int k0 = kt * 32 + ((lane & 16) ? 8 : 0);
      union { v16h v; v8h r[2]; } au;
      au.r[0] = load8_x1(p0row, v0row, k0);
      au.r[1] = load8_x1(p0row, v0row, k0 + 16);
      a[kt] = au.v;
    }
    int col0 = lane & 15;
#pragma unroll
    for (int nt = 0; nt < 8; ++nt) {
      v8f c = {};
#pragma unroll
      for (int kt = 0; kt < 4; ++kt) {
        v16h b = *(const v16h*)(w1frag + ((size_t)(nt * 4 + kt) * 32 + lane) * 16);
        c = __builtin_amdgcn_wmma_f32_16x16x32_f16(false, a[kt], false, b,
                                                   (short)0, c, false, false);
      }
      float ps = 0.0f, pq = 0.0f;
#pragma unroll
      for (int j = 0; j < 8; ++j) { ps += c[j]; pq += c[j] * c[j]; }
      int col = nt * 16 + col0;
      atomicAdd(&s[col], ps);
      atomicAdd(&s[128 + col], pq);
    }
  }
  __syncthreads();
  atomicAdd(&sums[t], s[t]);
}

// ---------------------------------------------------------------------------
// GEMM1 pass 2: recompute y1, BN+ReLU, atomic segment-max into d_out.
// ---------------------------------------------------------------------------
__global__ __launch_bounds__(256) void k_gemm1_apply(
    const h16_t* __restrict__ p0, const float* __restrict__ v0,
    const int* __restrict__ inv, const h16_t* __restrict__ w1frag,
    const float* __restrict__ scale, const float* __restrict__ shift,
    float* __restrict__ out, int n) {
  int t = threadIdx.x;
  int lane = t & 31;
  int rb = blockIdx.x * 8 + (t >> 5);
  if (rb * 16 >= n) return;
  int row = rb * 16 + (lane & 15);
  int vx = inv[row];
  const h16_t* p0row = p0 + (size_t)row * 64;
  const float* v0row = v0 + (size_t)vx * 64;
  v16h a[4];
#pragma unroll
  for (int kt = 0; kt < 4; ++kt) {
    int k0 = kt * 32 + ((lane & 16) ? 8 : 0);
    union { v16h v; v8h r[2]; } au;
    au.r[0] = load8_x1(p0row, v0row, k0);
    au.r[1] = load8_x1(p0row, v0row, k0 + 16);
    a[kt] = au.v;
  }
  int hi = (lane & 16) ? 8 : 0;
  int vox[8];
#pragma unroll
  for (int j = 0; j < 8; ++j) vox[j] = inv[rb * 16 + hi + j];
  int col0 = lane & 15;
#pragma unroll
  for (int nt = 0; nt < 8; ++nt) {
    v8f c = {};
#pragma unroll
    for (int kt = 0; kt < 4; ++kt) {
      v16h b = *(const v16h*)(w1frag + ((size_t)(nt * 4 + kt) * 32 + lane) * 16);
      c = __builtin_amdgcn_wmma_f32_16x16x32_f16(false, a[kt], false, b,
                                                 (short)0, c, false, false);
    }
    int col = nt * 16 + col0;
    float sc = scale[col], sh = shift[col];
#pragma unroll
    for (int j = 0; j < 8; ++j) {
      float val = fmaxf(c[j] * sc + sh, 0.0f);
      atomicMax((int*)&out[(size_t)vox[j] * 128 + col], __float_as_int(val));
    }
  }
}

// ---------------------------------------------------------------------------
// Host-side launch
// ---------------------------------------------------------------------------
extern "C" void kernel_launch(void* const* d_in, const int* in_sizes, int n_in,
                              void* d_out, int out_size, void* d_ws, size_t ws_size,
                              hipStream_t stream) {
  const float* features = (const float*)d_in[0];
  const float* w0 = (const float*)d_in[1];
  const float* g0 = (const float*)d_in[2];
  const float* b0 = (const float*)d_in[3];
  const float* w1 = (const float*)d_in[4];
  const float* g1 = (const float*)d_in[5];
  const float* b1 = (const float*)d_in[6];
  const int* coors = (const int*)d_in[7];
  const int* unq_inv = (const int*)d_in[8];
  float* out = (float*)d_out;

  const int N = in_sizes[0] / 4;      // 600000 points (multiple of 16)
  const int M = out_size / 128;       // 60000 voxels

  // ---- workspace carve-up (256B aligned) ----
  uint8_t* base = (uint8_t*)d_ws;
  size_t off = 0;
  auto carve = [&](size_t bytes) -> uint8_t* {
    uint8_t* p = base + off;
    off = (off + bytes + 255) & ~(size_t)255;
    return p;
  };
  float* vsum   = (float*)carve((size_t)M * 4 * sizeof(float));
  float* cnt    = (float*)carve((size_t)M * sizeof(float));
  h16_t* x10    = (h16_t*)carve((size_t)N * 16 * sizeof(h16_t));
  h16_t* p0     = (h16_t*)carve((size_t)N * 64 * sizeof(h16_t));
  float* v0     = (float*)carve((size_t)M * 64 * sizeof(float));
  h16_t* w0frag = (h16_t*)carve(4 * 512 * sizeof(h16_t));
  h16_t* w1frag = (h16_t*)carve(32 * 512 * sizeof(h16_t));
  float* sums0  = (float*)carve(128 * sizeof(float));
  float* sums1  = (float*)carve(256 * sizeof(float));
  float* scale0 = (float*)carve(64 * sizeof(float));
  float* shift0 = (float*)carve(64 * sizeof(float));
  float* scale1 = (float*)carve(128 * sizeof(float));
  float* shift1 = (float*)carve(128 * sizeof(float));
  (void)ws_size; (void)n_in;

  // ---- re-zero all accumulators every call (graph-replay safe) ----
  hipMemsetAsync(vsum, 0, (size_t)M * 4 * sizeof(float), stream);
  hipMemsetAsync(cnt, 0, (size_t)M * sizeof(float), stream);
  hipMemsetAsync(v0, 0, (size_t)M * 64 * sizeof(float), stream);
  hipMemsetAsync(sums0, 0, 128 * sizeof(float), stream);
  hipMemsetAsync(sums1, 0, 256 * sizeof(float), stream);
  hipMemsetAsync(out, 0, (size_t)out_size * sizeof(float), stream);

  const int ptBlocks = (N + 255) / 256;
  const int gBlocks = (N + 127) / 128;  // 8 waves/block, 16 rows/wave
  const float invN = 1.0f / (float)N;

  k_prep_weights<<<1, 256, 0, stream>>>(w0, w1, w0frag, w1frag);
  k_scatter_sum<<<ptBlocks, 256, 0, stream>>>(features, unq_inv, vsum, cnt, N);
  k_build_x10<<<ptBlocks, 256, 0, stream>>>(features, coors, unq_inv, vsum, cnt, x10, N);

  k_gemm0_stats<<<gBlocks, 256, 0, stream>>>(x10, w0frag, sums0, N);
  k_finalize_bn<<<1, 64, 0, stream>>>(sums0, g0, b0, scale0, shift0, 64, invN);
  k_gemm0_apply<<<gBlocks, 256, 0, stream>>>(x10, w0frag, scale0, shift0,
                                             unq_inv, p0, v0, N);

  k_gemm1_stats<<<gBlocks, 256, 0, stream>>>(p0, v0, unq_inv, w1frag, sums1, N);
  k_finalize_bn<<<1, 128, 0, stream>>>(sums1, g1, b1, scale1, shift1, 128, invN);
  k_gemm1_apply<<<gBlocks, 256, 0, stream>>>(p0, v0, unq_inv, w1frag,
                                             scale1, shift1, out, N);
}
